// HedgehogAttention_89343909692214
// MI455X (gfx1250) — matrix operand order repaired and checked by reading, not compiled, not run on hardware
//
#include <hip/hip_runtime.h>
#include <hip/hip_bf16.h>
#include <cstdint>

// MI455X / gfx1250, wave32. WMMA bf16: D(f32 16x16) = A(bf16 16x32) x B(bf16 32x16) + C.
typedef __attribute__((ext_vector_type(16))) __bf16 v16bf;
typedef __attribute__((ext_vector_type(8)))  __bf16 v8bf;
typedef __attribute__((ext_vector_type(8)))  float  v8f;

#define BATCH 2
#define SEQ   2048
#define DMODEL 1024
#define NHEAD 16
#define DHEAD 64
#define QKV_N (3 * DMODEL)   // 3072
#define QT 2                 // Q tiles per wave in attention

// ---------------------------------------------------------------------------
// Kernel 1: QKV = X @ W  (M=4096, K=1024, N=3072), bf16 out in [b][h][n][d].
// Block tile 128x128: 8 waves, wave w owns rows [16w,16w+16) and ALL 8
// col-subtiles (8 accumulators, 8 WMMAs per 32-K step from one shared W slab).
// Grid: (3072/128, 4096/128) = (24, 32).
// ---------------------------------------------------------------------------
__global__ __launch_bounds__(256) void qkv_gemm_kernel(
    const float* __restrict__ x, const float* __restrict__ w,
    __bf16* __restrict__ qw, __bf16* __restrict__ kw, __bf16* __restrict__ vw)
{
    __shared__ __align__(32) __bf16 wt[128 * 32];   // [col 0..127][k 0..31], 8 KB

    const int tid     = threadIdx.x;
    const int lane    = tid & 31;
    const int wave    = tid >> 5;
    const int colbase = blockIdx.x * 128;
    const int rowbase = blockIdx.y * 128;

    // A 16x32 layout: lane<16 -> row=lane, K 0..15; lane>=16 -> K 16..31
    const int arow  = lane & 15;
    const int khalf = (lane >> 4) * 16;
    // B 32x16 layout: col = lane&15, K = 16*(lane>=16) .. +15
    const int bks   = (lane >> 4) * 16;
    const int myrow = rowbase + wave * 16 + arow;

    v8f zero = {};
    v8f acc[8];
    #pragma unroll
    for (int t = 0; t < 8; ++t) acc[t] = zero;

    for (int k0 = 0; k0 < DMODEL; k0 += 32) {
        // Cooperative stage: W[k0..k0+31][colbase..+127] -> wt[col][k] bf16.
        #pragma unroll
        for (int i = 0; i < 2; ++i) {
            int cid  = tid * 2 + i;        // 0..511 chunks of 8 floats
            int krow = cid >> 4;           // 0..31
            int c8   = (cid & 15) * 8;     // 0..120
            v8f wv = *(const v8f*)&w[(size_t)(k0 + krow) * QKV_N + colbase + c8];
            #pragma unroll
            for (int j = 0; j < 8; ++j)
                wt[(c8 + j) * 32 + krow] = (__bf16)wv[j];
        }
        // Warm L2/WGP$ for the next W slab (global_prefetch_b8).
        if (k0 + 32 < DMODEL) {
            int cid  = tid * 2;
            int krow = cid >> 4;
            int c8   = (cid & 15) * 8;
            __builtin_prefetch(&w[(size_t)(k0 + 32 + krow) * QKV_N + colbase + c8], 0, 1);
        }
        __syncthreads();

        // A fragment from X (fp32 -> bf16), two contiguous 8-float chunks.
        const float* xr = &x[(size_t)myrow * DMODEL + k0 + khalf];
        v8f xlo = *(const v8f*)xr;
        v8f xhi = *(const v8f*)(xr + 8);
        v16bf a;
        #pragma unroll
        for (int j = 0; j < 8; ++j) { a[j] = (__bf16)xlo[j]; a[j + 8] = (__bf16)xhi[j]; }

        // 8 col-subtiles, one A reused across 8 WMMAs.
        #pragma unroll
        for (int t = 0; t < 8; ++t) {
            v16bf b = *(const v16bf*)&wt[(t * 16 + (lane & 15)) * 32 + bks];
            acc[t] = __builtin_amdgcn_wmma_f32_16x16x32_bf16(false, a, false, b,
                                                             (short)0, acc[t], false, false);
        }
        __syncthreads();
    }

    // Scatter C tiles -> bf16 q/k/v workspace in [b][h][n][d] layout.
    #pragma unroll
    for (int t = 0; t < 8; ++t) {
        #pragma unroll
        for (int r = 0; r < 8; ++r) {
            int m = rowbase + wave * 16 + r + ((lane >> 4) * 8);
            int c = colbase + t * 16 + (lane & 15);
            int which = c / DMODEL;
            int rem   = c % DMODEL;
            int h = rem / DHEAD;
            int d = rem % DHEAD;
            int bb = m / SEQ;
            int n  = m % SEQ;
            float val = acc[t][r];
            if (which == 0) val *= 0.125f;        // fold dhead^-0.5 into Q
            __bf16* dst = (which == 0) ? qw : (which == 1) ? kw : vw;
            dst[(((size_t)bb * NHEAD + h) * SEQ + n) * DHEAD + d] = (__bf16)val;
        }
    }
}

// ---------------------------------------------------------------------------
// Kernel 2: flash attention per (b,h). Grid: (SEQ/(16*8*QT), BATCH*NHEAD) =
// (8, 32). Block = 8 waves; wave owns QT=2 16-row Q tiles and loops over 64
// chunks of 32 K/V rows. K chunk staged via gfx1250 async-to-LDS block copy;
// V staged transposed through VGPRs (overlapped with the async copy).
// ---------------------------------------------------------------------------
__global__ __launch_bounds__(256) void attn_kernel(
    const __bf16* __restrict__ qw, const __bf16* __restrict__ kw,
    const __bf16* __restrict__ vw,
    const float* __restrict__ x, float* __restrict__ out)
{
    __shared__ __align__(32) __bf16 ldsK [32 * 64];      // [krow][d]   4 KB
    __shared__ __align__(32) __bf16 ldsVt[64 * 32];      // [d][krow]   4 KB
    __shared__ __align__(32) __bf16 ldsP [8 * 16 * 32];  // per-wave P  8 KB

    const int tid  = threadIdx.x;
    const int lane = tid & 31;
    const int wave = tid >> 5;
    const int bh   = blockIdx.y;
    const int bb   = bh >> 4;
    const int h    = bh & 15;
    const int qtile0 = (blockIdx.x * 8 + wave) * QT;
    const size_t base = (size_t)bh * SEQ * DHEAD;

    const int arow  = lane & 15;
    const int khalf = (lane >> 4) * 16;
    const int bcol  = lane & 15;
    const int bks   = (lane >> 4) * 16;
    __bf16* ldsPw = &ldsP[wave * 16 * 32];

    // Persistent Q A-fragments: QT tiles x d-chunks {[0,32),[32,64)}.
    v16bf qa[QT][2];
    #pragma unroll
    for (int qt = 0; qt < QT; ++qt) {
        int qbase = (qtile0 + qt) * 16;
        #pragma unroll
        for (int f = 0; f < 2; ++f) {
            const __bf16* qp = qw + base + (size_t)(qbase + arow) * DHEAD + f * 32 + khalf;
            v8bf lo = *(const v8bf*)qp;
            v8bf hi = *(const v8bf*)(qp + 8);
            #pragma unroll
            for (int j = 0; j < 8; ++j) { qa[qt][f][j] = lo[j]; qa[qt][f][j + 8] = hi[j]; }
        }
    }

    v8f zero = {};
    v8f o[QT][4];
    float m8[QT][8], l8[QT][8];
    #pragma unroll
    for (int qt = 0; qt < QT; ++qt) {
        #pragma unroll
        for (int g = 0; g < 4; ++g) o[qt][g] = zero;
        #pragma unroll
        for (int r = 0; r < 8; ++r) { m8[qt][r] = -3.0e38f; l8[qt][r] = 0.0f; }
    }

    for (int kb = 0; kb < SEQ; kb += 32) {
        // --- K chunk: linear 4 KB block copy via gfx1250 async-to-LDS. ---
        // dsaddr VGPR = workgroup-LDS byte offset (low 32 bits of flat ptr).
        {
            uint32_t ldsoff = (uint32_t)(uintptr_t)&ldsK[tid * 8];
            uint64_t ga = (uint64_t)(uintptr_t)(kw + base + (size_t)kb * DHEAD + tid * 8);
            asm volatile("global_load_async_to_lds_b128 %0, %1, off"
                         :: "v"(ldsoff), "v"(ga) : "memory");
        }
        // --- V chunk: transpose-stage through VGPRs (overlaps async copy). ---
        {
            int krow = tid >> 3;            // 0..31
            int d8   = (tid & 7) * 8;       // 0..56
            v8bf vv = *(const v8bf*)&vw[base + (size_t)(kb + krow) * DHEAD + d8];
            #pragma unroll
            for (int j = 0; j < 8; ++j)
                ldsVt[(d8 + j) * 32 + krow] = vv[j];
            if (kb + 32 < SEQ)
                __builtin_prefetch(&vw[base + (size_t)(kb + 32 + krow) * DHEAD + d8], 0, 1);
        }
        asm volatile("s_wait_asynccnt 0" ::: "memory");
        __syncthreads();

        #pragma unroll
        for (int qt = 0; qt < QT; ++qt) {
            // S = Q @ K^T : two 16x16 subtiles (k-cols [0,16) and [16,32)).
            v8f s[2];
            #pragma unroll
            for (int jt = 0; jt < 2; ++jt) {
                int kr = jt * 16 + bcol;
                v16bf b0 = *(const v16bf*)&ldsK[kr * 64 + bks];        // d [0,32)
                v16bf b1 = *(const v16bf*)&ldsK[kr * 64 + 32 + bks];   // d [32,64)
                v8f z = {};
                z = __builtin_amdgcn_wmma_f32_16x16x32_bf16(false, qa[qt][0], false, b0,
                                                            (short)0, z, false, false);
                z = __builtin_amdgcn_wmma_f32_16x16x32_bf16(false, qa[qt][1], false, b1,
                                                            (short)0, z, false, false);
                s[jt] = z;
            }

            // Online softmax. C layout: VGPR r, lane l -> row r+8*(l>=16), col l&15.
            #pragma unroll
            for (int r = 0; r < 8; ++r) {
                float mx = fmaxf(s[0][r], s[1][r]);
                mx = fmaxf(mx, __shfl_xor(mx, 1, 32));
                mx = fmaxf(mx, __shfl_xor(mx, 2, 32));
                mx = fmaxf(mx, __shfl_xor(mx, 4, 32));
                mx = fmaxf(mx, __shfl_xor(mx, 8, 32));
                float newm = fmaxf(m8[qt][r], mx);
                float sc   = __expf(m8[qt][r] - newm);
                m8[qt][r] = newm;
                float p0 = __expf(s[0][r] - newm);
                float p1 = __expf(s[1][r] - newm);
                float rs = p0 + p1;
                rs += __shfl_xor(rs, 1, 32);
                rs += __shfl_xor(rs, 2, 32);
                rs += __shfl_xor(rs, 4, 32);
                rs += __shfl_xor(rs, 8, 32);
                l8[qt][r] = l8[qt][r] * sc + rs;
                #pragma unroll
                for (int g = 0; g < 4; ++g) o[qt][g][r] *= sc;
                int prow = r + ((lane >> 4) * 8);
                ldsPw[prow * 32 + (lane & 15)]      = (__bf16)p0;
                ldsPw[prow * 32 + 16 + (lane & 15)] = (__bf16)p1;
            }
            // Intra-wave cross-lane LDS handoff: drain DS ops.
            asm volatile("s_wait_dscnt 0" ::: "memory");

            // P A-fragment (16x32 over this chunk's 32 k-values).
            v16bf pa;
            {
                const __bf16* pp = &ldsPw[arow * 32 + khalf];
                v8bf lo = *(const v8bf*)pp;
                v8bf hi = *(const v8bf*)(pp + 8);
                #pragma unroll
                for (int j = 0; j < 8; ++j) { pa[j] = lo[j]; pa[j + 8] = hi[j]; }
            }

            // O[g] += P @ V for d groups of 16.
            #pragma unroll
            for (int g = 0; g < 4; ++g) {
                v16bf vb = *(const v16bf*)&ldsVt[(g * 16 + bcol) * 32 + bks];
                o[qt][g] = __builtin_amdgcn_wmma_f32_16x16x32_bf16(false, pa, false, vb,
                                                                  (short)0, o[qt][g], false, false);
            }
        }
        __syncthreads();
    }

    // Epilogue: normalize, add residual, store fp32 [b][n][h*d].
    #pragma unroll
    for (int qt = 0; qt < QT; ++qt) {
        int qbase = (qtile0 + qt) * 16;
        #pragma unroll
        for (int g = 0; g < 4; ++g) {
            #pragma unroll
            for (int r = 0; r < 8; ++r) {
                int row = qbase + r + ((lane >> 4) * 8);
                int col = h * DHEAD + g * 16 + (lane & 15);
                size_t idx = ((size_t)bb * SEQ + row) * DMODEL + col;
                out[idx] = o[qt][g][r] / l8[qt][r] + x[idx];
            }
        }
    }
}

// ---------------------------------------------------------------------------
extern "C" void kernel_launch(void* const* d_in, const int* in_sizes, int n_in,
                              void* d_out, int out_size, void* d_ws, size_t ws_size,
                              hipStream_t stream)
{
    const float* x = (const float*)d_in[0];   // [2, 2048, 1024] f32
    const float* w = (const float*)d_in[1];   // [1024, 3072]   f32
    float* out = (float*)d_out;               // [2, 2048, 1024] f32

    // Workspace: q/k/v bf16, each [B][H][N][DH] = 4,194,304 elems (25.2 MB total).
    const size_t qelems = (size_t)BATCH * NHEAD * SEQ * DHEAD;
    __bf16* qw = (__bf16*)d_ws;
    __bf16* kw = qw + qelems;
    __bf16* vw = kw + qelems;

    dim3 g1(QKV_N / 128, (BATCH * SEQ) / 128);          // 24 x 32 blocks
    qkv_gemm_kernel<<<g1, 256, 0, stream>>>(x, w, qw, kw, vw);

    dim3 g2(SEQ / (16 * 8 * QT), BATCH * NHEAD);        // 8 x 32 blocks
    attn_kernel<<<g2, 256, 0, stream>>>(qw, kw, vw, x, out);
}
